// Model_41927470743971
// MI455X (gfx1250) — matrix operand order repaired and checked by reading, not compiled, not run on hardware
//
#include <hip/hip_runtime.h>

// ---------------------------------------------------------------------------
// BiLSTM-CRF (char BiLSTM + word BiLSTM + CRF NLL) for MI455X / gfx1250.
// Matrix work mapped to v_wmma_f32_16x16x32_f16 (f16 in, f32 accumulate).
// Strategy: precompute X@Wih for all timesteps (parallel WMMA GEMM), keep only
// H@Whh in the sequential recurrence (Whh is L2-resident). Gates padded to
// 112-wide blocks so i/f/g/o align per (VGPR,lane) across accumulators.
// ---------------------------------------------------------------------------

typedef _Float16 v16h __attribute__((ext_vector_type(16)));
typedef float    v8f  __attribute__((ext_vector_type(8)));

__device__ __forceinline__ v8f wmma16(v16h a, v16h b, v8f c) {
  // 8 args: (neg_a, A, neg_b, B, c_mod, C, reuse_a, reuse_b)
  return __builtin_amdgcn_wmma_f32_16x16x32_f16(false, a, false, b, (short)0, c, false, false);
}

__device__ __forceinline__ float sigm(float x) { return 1.f / (1.f + __expf(-x)); }

// A-matrix (16x32 f16) fragment: lane holds row m=lane&15; half h -> k per ISA layout.
__device__ __forceinline__ v16h load_a_frag(const _Float16* base, int stride, int lane, int kc) {
  const _Float16* row = base + (lane & 15) * stride;
  int kb = kc + ((lane >> 4) << 3);
  v16h a;
#pragma unroll
  for (int h = 0; h < 16; ++h) {
    int k = kb + ((h & 8) << 1) + (h & 7);
    a[h] = row[k];
  }
  return a;
}

__device__ __forceinline__ v16h load_a_frag_bounded(const _Float16* base, int stride, int lane,
                                                    int kc, int kmax) {
  const _Float16* row = base + (lane & 15) * stride;
  int kb = kc + ((lane >> 4) << 3);
  v16h a;
#pragma unroll
  for (int h = 0; h < 16; ++h) {
    int k = kb + ((h & 8) << 1) + (h & 7);
    a[h] = (k < kmax) ? row[k] : (_Float16)0.f;
  }
  return a;
}

// B-matrix (32x16 f16) fragment: lane holds col n=n0+(lane&15); half h -> k per ISA layout.
__device__ __forceinline__ v16h load_b_frag(const _Float16* base, int stride, int lane,
                                            int kc, int n0) {
  const _Float16* col = base + (size_t)kc * stride + n0 + (lane & 15);
  int kb = (lane >> 4) << 3;
  v16h b;
#pragma unroll
  for (int h = 0; h < 16; ++h) {
    int k = kb + ((h & 8) << 1) + (h & 7);
    b[h] = col[(size_t)k * stride];
  }
  return b;
}

// ======================= problem constants =======================
#define NB 64      // batch
#define NT 256     // seq len
#define TC 16      // char len
#define NW 16384   // NB*NT words
#define NL 25      // tags

// ======================= workspace layout =======================
constexpr size_t al256(size_t x) { return (x + 255) & ~(size_t)255; }
constexpr size_t OFF_CHARFEAT = 0;                               // f32 [16384][50]
constexpr size_t OFF_X16   = al256(OFF_CHARFEAT + (size_t)NW*50*4);      // f16 [16384][160]
constexpr size_t OFF_WIH   = al256(OFF_X16  + (size_t)NW*160*2);         // f16 [2][160][448]
constexpr size_t OFF_WHH   = al256(OFF_WIH  + (size_t)2*160*448*2);      // f16 [2][128][448]
constexpr size_t OFF_WBP   = al256(OFF_WHH  + (size_t)2*128*448*2);      // f32 [2][448]
constexpr size_t OFF_WTAG  = al256(OFF_WBP  + (size_t)2*448*4);          // f16 [224][32]
constexpr size_t OFF_GIH   = al256(OFF_WTAG + (size_t)224*32*2);         // f32 [2][16384][448]
constexpr size_t OFF_SEQ   = al256(OFF_GIH  + (size_t)2*NW*448*4);       // f16 [16384][224]
constexpr size_t OFF_EM    = al256(OFF_SEQ  + (size_t)NW*224*2);         // f32 [16384][25]
constexpr size_t OFF_NLL   = al256(OFF_EM   + (size_t)NW*25*4);          // f32 [64]

// ======================= kernel 0: pack word weights ========================
// Pad each 100-wide gate block to 112 (7 tiles of 16). Wih rows 150->160,
// Whh rows 100->128, W_tag 200x25 -> 224x32. All f16.
__global__ __launch_bounds__(256) void pack_weights_kernel(
    const float* wWih_f, const float* wWih_b,
    const float* wWhh_f, const float* wWhh_b,
    const float* wb_f, const float* wb_b, const float* W_tag,
    _Float16* Wih_pad, _Float16* Whh_pad, float* wb_pad, _Float16* Wtag_pad) {
  int idx = blockIdx.x * blockDim.x + threadIdx.x;
  const int R0 = 2 * 160 * 448, R1 = 2 * 128 * 448, R2 = 2 * 448, R3 = 224 * 32;
  if (idx < R0) {
    int dir = idx / (160 * 448);
    int r = idx % (160 * 448);
    int k = r / 448, pn = r % 448;
    int g = pn / 112, jj = pn % 112;
    const float* src = dir ? wWih_b : wWih_f;
    float v = (k < 150 && jj < 100) ? src[k * 400 + g * 100 + jj] : 0.f;
    Wih_pad[idx] = (_Float16)v;
    return;
  }
  idx -= R0;
  if (idx < R1) {
    int dir = idx / (128 * 448);
    int r = idx % (128 * 448);
    int k = r / 448, pn = r % 448;
    int g = pn / 112, jj = pn % 112;
    const float* src = dir ? wWhh_b : wWhh_f;
    float v = (k < 100 && jj < 100) ? src[k * 400 + g * 100 + jj] : 0.f;
    Whh_pad[idx] = (_Float16)v;
    return;
  }
  idx -= R1;
  if (idx < R2) {
    int dir = idx / 448, pn = idx % 448;
    int g = pn / 112, jj = pn % 112;
    const float* src = dir ? wb_b : wb_f;
    wb_pad[idx] = (jj < 100) ? src[g * 100 + jj] : 0.f;
    return;
  }
  idx -= R2;
  if (idx < R3) {
    int k = idx / 32, n = idx % 32;
    float v = (k < 200 && n < 25) ? W_tag[k * 25 + n] : 0.f;
    Wtag_pad[idx] = (_Float16)v;
  }
}

// ======================= kernel 1: char BiLSTM ========================
// One wave = 16 words, full 16-step LSTM, both directions sequentially.
// A = [x_t | h_{t-1}] padded to K=64; B = [Wih;Whh] padded [64][112] in LDS.
#define CWPT 4  // waves per block
__global__ __launch_bounds__(128) void char_lstm_kernel(
    const int* char_tensor, const int* char_lengths, const float* char_emb,
    const float* cWih_f, const float* cWhh_f, const float* cb_f,
    const float* cWih_b, const float* cWhh_b, const float* cb_b,
    float* char_feat) {
  __shared__ __align__(32) _Float16 sW[64][112];
  __shared__ float sBias[112];
  __shared__ __align__(32) _Float16 sA[CWPT][16][64];
  __shared__ float sGate[CWPT][16][112];
  __shared__ float sC[CWPT][16][25];
  __shared__ int sLen[CWPT][16];

  const int tid = threadIdx.x;
  const int lane = tid & 31;
  const int widx = tid >> 5;
  const int w0 = (blockIdx.x * CWPT + widx) * 16;
  const int mlo = (lane >> 4) << 3;

  for (int dir = 0; dir < 2; ++dir) {
    const float* Wih = dir ? cWih_b : cWih_f;
    const float* Whh = dir ? cWhh_b : cWhh_f;
    const float* bias = dir ? cb_b : cb_f;
    __syncthreads();
    // combined weights: rows 0..24 Wih, 25..49 Whh, rest zero; cols>=100 zero
    for (int i = tid; i < 64 * 112; i += 128) {
      int k = i / 112, n = i % 112;
      float v = 0.f;
      if (n < 100) {
        if (k < 25) v = Wih[k * 100 + n];
        else if (k < 50) v = Whh[(k - 25) * 100 + n];
      }
      sW[k][n] = (_Float16)v;
    }
    for (int i = tid; i < 112; i += 128) sBias[i] = (i < 100) ? bias[i] : 0.f;
    for (int i = lane; i < 16 * 64; i += 32) sA[widx][i >> 6][i & 63] = (_Float16)0.f;
    for (int i = lane; i < 16 * 25; i += 32) sC[widx][i / 25][i % 25] = 0.f;
    if (lane < 16) sLen[widx][lane] = char_lengths[w0 + lane];
    __syncthreads();

    for (int t = 0; t < TC; ++t) {
      // gather x_t (reversed within length for the backward direction)
      for (int i = lane; i < 16 * 25; i += 32) {
        int wl = i / 25, j = i % 25;
        int tt = t;
        if (dir) { int L = sLen[widx][wl]; tt = (t < L) ? (L - 1 - t) : t; }
        int ci = char_tensor[(w0 + wl) * TC + tt];
        sA[widx][wl][j] = (_Float16)char_emb[ci * 25 + j];
      }
      __syncthreads();

      v16h a0 = load_a_frag(&sA[widx][0][0], 64, lane, 0);
      v16h a1 = load_a_frag(&sA[widx][0][0], 64, lane, 32);
      v8f acc[7];
#pragma unroll
      for (int nt = 0; nt < 7; ++nt) {
        float bv = sBias[nt * 16 + (lane & 15)];
#pragma unroll
        for (int r = 0; r < 8; ++r) acc[nt][r] = bv;
        acc[nt] = wmma16(a0, load_b_frag(&sW[0][0], 112, lane, 0, nt * 16), acc[nt]);
        acc[nt] = wmma16(a1, load_b_frag(&sW[0][0], 112, lane, 32, nt * 16), acc[nt]);
      }
#pragma unroll
      for (int nt = 0; nt < 7; ++nt) {
#pragma unroll
        for (int r = 0; r < 8; ++r) {
          sGate[widx][r + mlo][nt * 16 + (lane & 15)] = acc[nt][r];
        }
      }
      __syncthreads();
      // pointwise LSTM update; capture last-valid hidden
      for (int i = lane; i < 16 * 25; i += 32) {
        int wl = i / 25, j = i % 25;
        float gi = sGate[widx][wl][j];
        float gf = sGate[widx][wl][25 + j];
        float gg = sGate[widx][wl][50 + j];
        float go = sGate[widx][wl][75 + j];
        float c = sigm(gf) * sC[widx][wl][j] + sigm(gi) * tanhf(gg);
        float h = sigm(go) * tanhf(c);
        sC[widx][wl][j] = c;
        sA[widx][wl][25 + j] = (_Float16)h;
        if (t == sLen[widx][wl] - 1)
          char_feat[(size_t)(w0 + wl) * 50 + dir * 25 + j] = h;
      }
      __syncthreads();
    }
  }
}

// ======================= kernel 2: build word-LSTM input ========================
// x[n][0:100]=word_emb[tok[n]], x[n][100:150]=char_feat[recover[n]], pad to 160. f16.
__global__ __launch_bounds__(256) void build_x_kernel(
    const int* tok, const int* recover, const float* word_emb,
    const float* char_feat, _Float16* x16) {
  int idx = blockIdx.x * blockDim.x + threadIdx.x;
  if (idx >= NW * 160) return;
  int n = idx / 160, col = idx % 160;
  float v = 0.f;
  if (col < 100) v = word_emb[(size_t)tok[n] * 100 + col];
  else if (col < 150) v = char_feat[(size_t)recover[n] * 50 + (col - 100)];
  x16[idx] = (_Float16)v;
}

// ======================= kernel 3: input-side GEMM (all timesteps) ==============
// G_ih[dir][n][448] = x16[n][0:160] @ Wih_pad[dir] + bias. One wave per (Mtile,Ntile,dir).
__global__ __launch_bounds__(256) void gemm_ih_kernel(
    const _Float16* x16, const _Float16* Wih_pad, const float* wb_pad, float* G_ih) {
  int lane = threadIdx.x & 31;
  int wid = blockIdx.x * 8 + (threadIdx.x >> 5);  // [0, 2*1024*28)
  int dir = wid / (1024 * 28);
  int rem = wid % (1024 * 28);
  int mt = rem / 28, nt = rem % 28;
  const _Float16* wih = Wih_pad + (size_t)dir * 160 * 448;
  int n0 = nt * 16;
  int n = n0 + (lane & 15);
  float bv = wb_pad[dir * 448 + n];
  v8f acc;
#pragma unroll
  for (int r = 0; r < 8; ++r) acc[r] = bv;
#pragma unroll
  for (int kc = 0; kc < 5; ++kc) {
    v16h a = load_a_frag(x16 + (size_t)mt * 16 * 160, 160, lane, kc * 32);
    v16h b = load_b_frag(wih, 448, lane, kc * 32, n0);
    acc = wmma16(a, b, acc);
  }
  int mlo = (lane >> 4) << 3;
  float* gout = G_ih + (size_t)dir * NW * 448;
#pragma unroll
  for (int r = 0; r < 8; ++r)
    gout[(size_t)(mt * 16 + r + mlo) * 448 + n] = acc[r];
}

// ======================= kernel 4: word LSTM recurrence ========================
// grid.x = 2 (fwd/bwd run concurrently). 4 waves = 4 batch tiles of 16.
// Per step: gates = G_ih[:,t] + H @ Whh; gate g lives in tiles g*7..g*7+6 so
// i/f/g/o for one unit share (VGPR,lane) -> pointwise math fully in registers.
__global__ __launch_bounds__(128) void word_recur_kernel(
    const float* G_ih, const _Float16* Whh_pad, _Float16* seq16) {
  __shared__ __align__(32) _Float16 sH[64][128];
  const int dir = blockIdx.x;
  const int lane = threadIdx.x & 31;
  const int widx = threadIdx.x >> 5;
  const int b0 = widx * 16;
  const int mlo = (lane >> 4) << 3;

  for (int i = threadIdx.x; i < 64 * 128; i += 128) sH[i / 128][i % 128] = (_Float16)0.f;
  __syncthreads();

  const float* gih = G_ih + (size_t)dir * NW * 448;
  const _Float16* whh = Whh_pad + (size_t)dir * 128 * 448;

  float cst[7][8];
#pragma unroll
  for (int tt = 0; tt < 7; ++tt)
#pragma unroll
    for (int r = 0; r < 8; ++r) cst[tt][r] = 0.f;

  for (int s = 0; s < NT; ++s) {
    int t = dir ? (NT - 1 - s) : s;
    v16h a[4];
#pragma unroll
    for (int kc = 0; kc < 4; ++kc) a[kc] = load_a_frag(&sH[b0][0], 128, lane, kc * 32);

#pragma unroll
    for (int tt = 0; tt < 7; ++tt) {
      v8f acc[4];
#pragma unroll
      for (int g = 0; g < 4; ++g) {
        int nt = g * 7 + tt;
        int n = nt * 16 + (lane & 15);
#pragma unroll
        for (int r = 0; r < 8; ++r) {
          int bb = b0 + r + mlo;
          acc[g][r] = gih[((size_t)bb * NT + t) * 448 + n];
        }
#pragma unroll
        for (int kc = 0; kc < 4; ++kc)
          acc[g] = wmma16(a[kc], load_b_frag(whh, 448, lane, kc * 32, nt * 16), acc[g]);
      }
      int j = tt * 16 + (lane & 15);
#pragma unroll
      for (int r = 0; r < 8; ++r) {
        if (j < 100) {
          float c = sigm(acc[1][r]) * cst[tt][r] + sigm(acc[0][r]) * tanhf(acc[2][r]);
          float h = sigm(acc[3][r]) * tanhf(c);
          cst[tt][r] = c;
          int bb = b0 + r + mlo;
          sH[bb][j] = (_Float16)h;
          seq16[((size_t)bb * NT + t) * 224 + dir * 100 + j] = (_Float16)h;
        }
      }
    }
    __syncthreads();
  }
}

// ======================= kernel 5: emissions ========================
// em[n][25] = seq16[n][0:200] @ Wtag_pad + b_tag  (K padded to 224, N to 32)
__global__ __launch_bounds__(256) void emissions_kernel(
    const _Float16* seq16, const _Float16* Wtag_pad, const float* b_tag, float* em) {
  int lane = threadIdx.x & 31;
  int wid = blockIdx.x * 8 + (threadIdx.x >> 5);  // [0, 1024*2)
  int mt = wid >> 1, nt = wid & 1;
  int n0 = nt * 16;
  int n = n0 + (lane & 15);
  float bv = (n < 25) ? b_tag[n] : 0.f;
  v8f acc;
#pragma unroll
  for (int r = 0; r < 8; ++r) acc[r] = bv;
#pragma unroll
  for (int kc = 0; kc < 7; ++kc) {
    v16h a = load_a_frag_bounded(seq16 + (size_t)mt * 16 * 224, 224, lane, kc * 32, 200);
    v16h b = load_b_frag(Wtag_pad, 32, lane, kc * 32, n0);
    acc = wmma16(a, b, acc);
  }
  if (n < 25) {
    int mlo = (lane >> 4) << 3;
#pragma unroll
    for (int r = 0; r < 8; ++r)
      em[(size_t)(mt * 16 + r + mlo) * 25 + n] = acc[r];
  }
}

// ======================= kernel 6: CRF NLL per batch element ====================
// One wave per batch row; lane = tag for the forward algorithm (L=25 <= 32).
__global__ __launch_bounds__(256) void crf_kernel(
    const float* em, const int* tags, const float* trans,
    const float* start, const float* end_v, float* nll_part) {
  __shared__ float sTr[25][25];
  __shared__ float sAl[8][32];
  __shared__ float sRed[8][32];
  const int lane = threadIdx.x & 31;
  const int widx = threadIdx.x >> 5;
  const int b = blockIdx.x * 8 + widx;
  for (int i = threadIdx.x; i < 625; i += 256) sTr[i / 25][i % 25] = trans[i];
  __syncthreads();

  const float* e = em + (size_t)b * NT * NL;
  const int* tg = tags + (size_t)b * NT;

  // gold path score (parallel over lanes, reduce via LDS)
  float part = 0.f;
  for (int t = lane; t < NT; t += 32) {
    int y = tg[t];
    part += e[t * NL + y];
    if (t > 0) part += sTr[tg[t - 1]][y];
  }
  sRed[widx][lane] = part;
  __syncthreads();
  float gold = 0.f;
  if (lane == 0) {
    float s = 0.f;
    for (int i = 0; i < 32; ++i) s += sRed[widx][i];
    gold = s + start[tg[0]] + end_v[tg[NT - 1]];
  }

  // forward algorithm
  float alpha = (lane < NL) ? (start[lane] + e[lane]) : -1e30f;
  for (int t = 1; t < NT; ++t) {
    sAl[widx][lane] = alpha;
    __syncthreads();
    float na = -1e30f;
    if (lane < NL) {
      float mx = -1e30f;
      for (int i = 0; i < NL; ++i) mx = fmaxf(mx, sAl[widx][i] + sTr[i][lane]);
      float s = 0.f;
      for (int i = 0; i < NL; ++i) s += __expf(sAl[widx][i] + sTr[i][lane] - mx);
      na = e[t * NL + lane] + mx + __logf(s);
    }
    __syncthreads();
    alpha = na;
  }
  sAl[widx][lane] = (lane < NL) ? (alpha + end_v[lane]) : -1e30f;
  __syncthreads();
  if (lane == 0) {
    float mx = -1e30f;
    for (int i = 0; i < NL; ++i) mx = fmaxf(mx, sAl[widx][i]);
    float s = 0.f;
    for (int i = 0; i < NL; ++i) s += __expf(sAl[widx][i] - mx);
    nll_part[b] = (mx + __logf(s)) - gold;
  }
}

__global__ void reduce_kernel(const float* nll_part, float* out) {
  if (threadIdx.x == 0) {
    float s = 0.f;
    for (int i = 0; i < NB; ++i) s += nll_part[i];
    out[0] = s;
  }
}

// ======================= launch ========================
extern "C" void kernel_launch(void* const* d_in, const int* in_sizes, int n_in,
                              void* d_out, int out_size, void* d_ws, size_t ws_size,
                              hipStream_t stream) {
  const int* tok        = (const int*)d_in[0];
  const int* tags       = (const int*)d_in[1];
  // d_in[2] lengths: full-length sequences, unused
  const int* char_t     = (const int*)d_in[3];
  const int* char_len   = (const int*)d_in[4];
  const int* recover    = (const int*)d_in[5];
  const float* word_emb = (const float*)d_in[6];
  const float* char_emb = (const float*)d_in[7];
  const float* cWih_f   = (const float*)d_in[8];
  const float* cWhh_f   = (const float*)d_in[9];
  const float* cb_f     = (const float*)d_in[10];
  const float* cWih_b   = (const float*)d_in[11];
  const float* cWhh_b   = (const float*)d_in[12];
  const float* cb_b     = (const float*)d_in[13];
  const float* wWih_f   = (const float*)d_in[14];
  const float* wWhh_f   = (const float*)d_in[15];
  const float* wb_f     = (const float*)d_in[16];
  const float* wWih_b   = (const float*)d_in[17];
  const float* wWhh_b   = (const float*)d_in[18];
  const float* wb_b     = (const float*)d_in[19];
  const float* W_tag    = (const float*)d_in[20];
  const float* b_tag    = (const float*)d_in[21];
  const float* trans    = (const float*)d_in[22];
  const float* start    = (const float*)d_in[23];
  const float* end_v    = (const float*)d_in[24];

  char* ws = (char*)d_ws;
  float*     char_feat = (float*)(ws + OFF_CHARFEAT);
  _Float16*  x16       = (_Float16*)(ws + OFF_X16);
  _Float16*  Wih_pad   = (_Float16*)(ws + OFF_WIH);
  _Float16*  Whh_pad   = (_Float16*)(ws + OFF_WHH);
  float*     wb_pad    = (float*)(ws + OFF_WBP);
  _Float16*  Wtag_pad  = (_Float16*)(ws + OFF_WTAG);
  float*     G_ih      = (float*)(ws + OFF_GIH);
  _Float16*  seq16     = (_Float16*)(ws + OFF_SEQ);
  float*     em        = (float*)(ws + OFF_EM);
  float*     nll_part  = (float*)(ws + OFF_NLL);

  pack_weights_kernel<<<1040, 256, 0, stream>>>(
      wWih_f, wWih_b, wWhh_f, wWhh_b, wb_f, wb_b, W_tag,
      Wih_pad, Whh_pad, wb_pad, Wtag_pad);
  char_lstm_kernel<<<NW / (16 * CWPT), 128, 0, stream>>>(
      char_t, char_len, char_emb, cWih_f, cWhh_f, cb_f, cWih_b, cWhh_b, cb_b, char_feat);
  build_x_kernel<<<(NW * 160 + 255) / 256, 256, 0, stream>>>(
      tok, recover, word_emb, char_feat, x16);
  gemm_ih_kernel<<<(2 * 1024 * 28) / 8, 256, 0, stream>>>(x16, Wih_pad, wb_pad, G_ih);
  word_recur_kernel<<<2, 128, 0, stream>>>(G_ih, Whh_pad, seq16);
  emissions_kernel<<<(1024 * 2) / 8, 256, 0, stream>>>(seq16, Wtag_pad, b_tag, em);
  crf_kernel<<<NB / 8, 256, 0, stream>>>(em, tags, trans, start, end_v, nll_part);
  reduce_kernel<<<1, 64, 0, stream>>>(nll_part, (float*)d_out);
}
